// Spreader_75179107549637
// MI455X (gfx1250) — compile-verified, matrix operand order checked by reading, not run on hardware
//
#include <hip/hip_runtime.h>
#include <math.h>

// ---------- types ----------
typedef __bf16 bf16;
typedef __attribute__((ext_vector_type(16))) __bf16 bf16x16;
typedef __attribute__((ext_vector_type(8)))  __bf16 bf16x8;
typedef __attribute__((ext_vector_type(8)))  float  f32x8;

// ---------- problem constants ----------
#define BB 8
#define SS 2048
#define DD 1024
#define MM (BB * SS)      // 16384 rows
#define EPSLN 1e-6f
#define KSTART 2.0f
#define KEND   8.0f
#define FACTOR 0.5f

// async copy: 16B per lane, global -> LDS, tracked by ASYNCcnt
__device__ __forceinline__ void async_g2l_b128(unsigned lds_off, const void* gptr) {
    asm volatile("global_load_async_to_lds_b128 %0, %1, off"
                 :: "v"(lds_off), "v"(gptr) : "memory");
}
__device__ __forceinline__ void wait_asynccnt0() {
    asm volatile("s_wait_asynccnt 0x0" ::: "memory");
}
__device__ __forceinline__ unsigned lds_addr_of(const void* p) {
    // generic pointers to LDS carry the LDS byte offset in the low 32 bits
    return (unsigned)(uintptr_t)p;
}

// =====================================================================
// One-time: decay[h] = 0.5^(1/k_h), dbeta[h] = 1 - decay[h]
// =====================================================================
__global__ void decay_init_kernel(float* __restrict__ decay, float* __restrict__ dbeta) {
    int h = blockIdx.x * 256 + threadIdx.x;
    if (h < DD) {
        float kk = KSTART + (KEND - KSTART) * ((float)h / (float)(DD - 1));
        float d = __powf(FACTOR, 1.0f / kk);
        decay[h] = d;
        dbeta[h] = 1.0f - d;
    }
}

// =====================================================================
// Weight convert + transpose:  out[n*Krows + k] = (bf16) in[k*N + n]
// =====================================================================
__global__ void wconv_t_kernel(const float* __restrict__ in, bf16* __restrict__ out,
                               int Krows, int Ncols) {
    int k = blockIdx.x * 16 + threadIdx.x;
    int n = blockIdx.y * 16 + threadIdx.y;
    if (k < Krows && n < Ncols) {
        out[(size_t)n * Krows + k] = (bf16)in[(size_t)k * Ncols + n];
    }
}

// =====================================================================
// Row LayerNorm over D=1024, one block (256 thr) per row.
// =====================================================================
__global__ __launch_bounds__(256)
void ln_row_kernel(const float* __restrict__ x,
                   const float* __restrict__ w, const float* __restrict__ b,
                   float* __restrict__ of, bf16* __restrict__ ob) {
    __shared__ float s_sum[8], s_sq[8];
    const size_t base = (size_t)blockIdx.x * DD;
    const int tid = threadIdx.x;

    float v[4];
    float sum = 0.f, sq = 0.f;
#pragma unroll
    for (int i = 0; i < 4; ++i) {
        v[i] = x[base + tid + i * 256];
        sum += v[i];
        sq  += v[i] * v[i];
    }
#pragma unroll
    for (int off = 16; off > 0; off >>= 1) {
        sum += __shfl_xor(sum, off, 32);
        sq  += __shfl_xor(sq,  off, 32);
    }
    if ((tid & 31) == 0) { s_sum[tid >> 5] = sum; s_sq[tid >> 5] = sq; }
    __syncthreads();
    float ts = 0.f, tq = 0.f;
#pragma unroll
    for (int i = 0; i < 8; ++i) { ts += s_sum[i]; tq += s_sq[i]; }
    const float mean = ts * (1.0f / DD);
    const float var  = tq * (1.0f / DD) - mean * mean;
    const float rstd = rsqrtf(var + EPSLN);
#pragma unroll
    for (int i = 0; i < 4; ++i) {
        int j = tid + i * 256;
        float y = (v[i] - mean) * rstd * w[j] + b[j];
        if (of) of[base + j] = y;
        if (ob) ob[base + j] = (bf16)y;
    }
}

// =====================================================================
// Decay scan along S (in-place):  cx[t] = decay[h]*cx[t-1] + u[t]
// =====================================================================
__global__ __launch_bounds__(256)
void scan_kernel(float* __restrict__ u, const float* __restrict__ decay_tab) {
    int t = blockIdx.x * 256 + threadIdx.x;   // 0 .. B*H-1
    int bb = t / DD, h = t % DD;
    float decay = decay_tab[h];
    float c = 0.f;
    size_t idx = (size_t)bb * SS * DD + h;
    for (int s = 0; s < SS; ++s) {
        c = decay * c + u[idx];
        u[idx] = c;
        idx += DD;
    }
}

// =====================================================================
// WMMA bf16 GEMM:  C[M,N] = A[M,K] * Bt[N,K]^T
// 256 thr = 8 waves; block tile 128x128; wave tile 32x64 (2x4 WMMA);
// K-chunk 32; double-buffered LDS fed by ASYNC global->LDS b128 copies.
// A source switches A1->A2 at K1 (concat GEMM without materializing).
// MODE 0: scale by tab[n] (decay_beta), store f32
// MODE 1: store f32 + bf16
// MODE 2: add tab[n] (bias), store f32
// =====================================================================
template <int MODE>
__global__ __launch_bounds__(256)
void gemm_wmma_kernel(const bf16* __restrict__ A1, const bf16* __restrict__ A2,
                      int K1, int K,
                      const bf16* __restrict__ Bt,
                      const float* __restrict__ tab,
                      float* __restrict__ Cf, bf16* __restrict__ Cb,
                      int N) {
    __shared__ __align__(16) bf16 As[2][128][40];   // padded rows (80B) vs banks
    __shared__ __align__(16) bf16 Bs[2][128][40];   // Bs[buf][n][k]

    const int tid  = threadIdx.x;
    const int wave = tid >> 5;
    const int lane = tid & 31;
    const int half = lane >> 4;       // K-half per ISA 16-bit A/B layout
    const int r    = lane & 15;

    const int m0 = blockIdx.y * 128;
    const int n0 = blockIdx.x * 128;
    const int wm = (wave >> 1) * 32;  // 0,32,64,96
    const int wn = (wave & 1) * 64;   // 0,64

    f32x8 acc[2][4] = {};

    auto aptr = [&](int k0) -> const bf16* {
        if (k0 < K1) return &A1[(size_t)m0 * K1 + k0];
        return &A2[(size_t)m0 * K1 + (k0 - K1)];
    };

    // stage one 128x32 A tile + 128x32 B tile into LDS buffer `buf`
    // via async b128 copies (4 per thread; 8 waves x 4 x 512B = 16KB)
    auto stage_async = [&](int k0, int buf) {
        const bf16* Ab = aptr(k0);
        const bf16* Bb = &Bt[(size_t)n0 * K + k0];
#pragma unroll
        for (int it = 0; it < 2; ++it) {
            int chunk = tid + it * 256;
            int row = chunk >> 2, c8 = chunk & 3;
            async_g2l_b128(lds_addr_of(&As[buf][row][c8 * 8]),
                           &Ab[(size_t)row * K1 + c8 * 8]);
            async_g2l_b128(lds_addr_of(&Bs[buf][row][c8 * 8]),
                           &Bb[(size_t)row * K + c8 * 8]);
        }
    };

    stage_async(0, 0);
    wait_asynccnt0();
    __syncthreads();

    for (int k0 = 0; k0 < K; k0 += 32) {
        const int buf = (k0 >> 5) & 1;
        const bool more = (k0 + 32) < K;

        if (more) stage_async(k0 + 32, buf ^ 1);   // overlap with WMMA below

        bf16x16 afrag[2], bfrag[4];
#pragma unroll
        for (int mi = 0; mi < 2; ++mi) {
            const bf16* p = &As[buf][wm + mi * 16 + r][half * 8];
            bf16x8 lo = *(const bf16x8*)p;
            bf16x8 hi = *(const bf16x8*)(p + 16);
            afrag[mi] = __builtin_shufflevector(lo, hi, 0,1,2,3,4,5,6,7,8,9,10,11,12,13,14,15);
        }
#pragma unroll
        for (int ni = 0; ni < 4; ++ni) {
            const bf16* p = &Bs[buf][wn + ni * 16 + r][half * 8];
            bf16x8 lo = *(const bf16x8*)p;
            bf16x8 hi = *(const bf16x8*)(p + 16);
            bfrag[ni] = __builtin_shufflevector(lo, hi, 0,1,2,3,4,5,6,7,8,9,10,11,12,13,14,15);
        }

#pragma unroll
        for (int mi = 0; mi < 2; ++mi)
#pragma unroll
            for (int ni = 0; ni < 4; ++ni)
                acc[mi][ni] = __builtin_amdgcn_wmma_f32_16x16x32_bf16(
                    false, afrag[mi], false, bfrag[ni],
                    (short)0, acc[mi][ni], false, false);

        if (more) {
            wait_asynccnt0();     // own async copies into buf^1 done
            __syncthreads();      // everyone's copies visible
        }
    }

    // epilogue: f32 C/D layout -> VGPR i = row (half*8 + i), col r
#pragma unroll
    for (int ni = 0; ni < 4; ++ni) {
        const int nn = n0 + wn + ni * 16 + r;
        float scale = 1.0f, add = 0.0f;
        if (MODE == 0) scale = tab[nn];
        if (MODE == 2) add = tab[nn];
#pragma unroll
        for (int mi = 0; mi < 2; ++mi) {
#pragma unroll
            for (int i = 0; i < 8; ++i) {
                int mm = m0 + wm + mi * 16 + half * 8 + i;
                float v = acc[mi][ni][i] * scale + add;
                size_t o = (size_t)mm * N + nn;
                Cf[o] = v;
                if (MODE == 1) Cb[o] = (bf16)v;
            }
        }
    }
}

// =====================================================================
// Final: gate = sigmoid(LN(gpre)); out = (1-g)*res + g*o
// =====================================================================
__global__ __launch_bounds__(256)
void final_kernel(const float* __restrict__ gpre,
                  const float* __restrict__ w, const float* __restrict__ b,
                  const float* __restrict__ res, const float* __restrict__ o,
                  float* __restrict__ out) {
    __shared__ float s_sum[8], s_sq[8];
    const size_t base = (size_t)blockIdx.x * DD;
    const int tid = threadIdx.x;
    float v[4];
    float sum = 0.f, sq = 0.f;
#pragma unroll
    for (int i = 0; i < 4; ++i) {
        v[i] = gpre[base + tid + i * 256];
        sum += v[i]; sq += v[i] * v[i];
    }
#pragma unroll
    for (int off = 16; off > 0; off >>= 1) {
        sum += __shfl_xor(sum, off, 32);
        sq  += __shfl_xor(sq,  off, 32);
    }
    if ((tid & 31) == 0) { s_sum[tid >> 5] = sum; s_sq[tid >> 5] = sq; }
    __syncthreads();
    float ts = 0.f, tq = 0.f;
#pragma unroll
    for (int i = 0; i < 8; ++i) { ts += s_sum[i]; tq += s_sq[i]; }
    const float mean = ts * (1.0f / DD);
    const float var  = tq * (1.0f / DD) - mean * mean;
    const float rstd = rsqrtf(var + EPSLN);
#pragma unroll
    for (int i = 0; i < 4; ++i) {
        int j = tid + i * 256;
        float ln = (v[i] - mean) * rstd * w[j] + b[j];
        float g = 1.0f / (1.0f + __expf(-ln));
        float rr = res[base + j], oo = o[base + j];
        out[base + j] = (1.0f - g) * rr + g * oo;
    }
}

// =====================================================================
extern "C" void kernel_launch(void* const* d_in, const int* in_sizes, int n_in,
                              void* d_out, int out_size, void* d_ws, size_t ws_size,
                              hipStream_t stream) {
    (void)in_sizes; (void)n_in; (void)out_size; (void)ws_size;

    const float* x     = (const float*)d_in[0];
    const float* Wt    = (const float*)d_in[1];
    const float* Wo    = (const float*)d_in[2];
    const float* Wg    = (const float*)d_in[3];
    const float* bg    = (const float*)d_in[4];
    const float* ln1_w = (const float*)d_in[5];
    const float* ln1_b = (const float*)d_in[6];
    const float* lnc_w = (const float*)d_in[7];
    const float* lnc_b = (const float*)d_in[8];
    const float* lng_w = (const float*)d_in[9];
    const float* lng_b = (const float*)d_in[10];
    float* out = (float*)d_out;

    // workspace carve-up (256B aligned)
    size_t off = 0;
    auto alloc = [&](size_t bytes) -> void* {
        off = (off + 255) & ~(size_t)255;
        void* p = (char*)d_ws + off;
        off += bytes;
        return p;
    };
    const size_t NE = (size_t)MM * DD;          // 16.78M elements
    float* xn    = (float*)alloc(NE * 4);       // LN1(x) f32 (residual)
    float* u     = (float*)alloc(NE * 4);       // u -> cx -> gpre (reused)
    float* outf  = (float*)alloc(NE * 4);       // out f32
    bf16*  xb    = (bf16*)alloc(NE * 2);        // LN1(x) bf16
    bf16*  cb    = (bf16*)alloc(NE * 2);        // LN(cx) bf16
    bf16*  outb  = (bf16*)alloc(NE * 2);        // out bf16
    bf16*  WtT   = (bf16*)alloc((size_t)DD * DD * 2);
    bf16*  WoT   = (bf16*)alloc((size_t)DD * DD * 2);
    bf16*  WgT   = (bf16*)alloc((size_t)2 * DD * DD * 2);
    float* decay = (float*)alloc(DD * 4);
    float* dbeta = (float*)alloc(DD * 4);

    decay_init_kernel<<<(DD + 255) / 256, 256, 0, stream>>>(decay, dbeta);

    dim3 t16(16, 16);
    wconv_t_kernel<<<dim3(DD / 16, DD / 16), t16, 0, stream>>>(Wt, WtT, DD, DD);
    wconv_t_kernel<<<dim3(DD / 16, DD / 16), t16, 0, stream>>>(Wo, WoT, DD, DD);
    wconv_t_kernel<<<dim3(2 * DD / 16, DD / 16), t16, 0, stream>>>(Wg, WgT, 2 * DD, DD);

    // LN1 -> xn (f32) + xb (bf16)
    ln_row_kernel<<<MM, 256, 0, stream>>>(x, ln1_w, ln1_b, xn, xb);

    dim3 ggrid(DD / 128, MM / 128);   // (8, 128)
    // u = LN1(x) @ Wt * decay_beta
    gemm_wmma_kernel<0><<<ggrid, 256, 0, stream>>>(xb, xb, DD, DD, WtT,
                                                   dbeta, u, nullptr, DD);
    // cx = decay scan (in-place on u)
    scan_kernel<<<(BB * DD) / 256, 256, 0, stream>>>(u, decay);
    // LN(cx) -> cb (bf16 only)
    ln_row_kernel<<<MM, 256, 0, stream>>>(u, lnc_w, lnc_b, nullptr, cb);
    // out = LN(cx) @ Wo  (f32 + bf16)
    gemm_wmma_kernel<1><<<ggrid, 256, 0, stream>>>(cb, cb, DD, DD, WoT,
                                                   nullptr, outf, outb, DD);
    // gpre = concat(xn_bf16, out_bf16) @ Wg + bg   (K = 2048; reuse u)
    gemm_wmma_kernel<2><<<ggrid, 256, 0, stream>>>(xb, outb, DD, 2 * DD, WgT,
                                                   bg, u, nullptr, DD);
    // gate + blend
    final_kernel<<<MM, 256, 0, stream>>>(u, lng_w, lng_b, xn, outf, out);
}
